// BasicTransformerBlock_1047972020813
// MI455X (gfx1250) — compile-verified, matrix operand order checked by reading, not run on hardware
//
#include <hip/hip_runtime.h>
#include <hip/hip_bf16.h>
#include <math.h>

// ---------------------------------------------------------------------------
// CDNA5 (gfx1250) BasicTransformerBlock. All GEMMs + attention matmuls via
// v_wmma_f32_16x16x32_f16. Matmul-input intermediates (LN outputs, Q/K/V,
// attention outputs, GEGLU inner) are stored as f16 in memory: they are
// f16 WMMA operands anyway, so this halves their HBM traffic. Residual
// stream stays f32. Fragment-ordered LDS tiles (2x ds_load_b128/operand),
// double-buffered GEMM, sparse-causal gather fused into attention staging.
// ---------------------------------------------------------------------------

typedef __attribute__((ext_vector_type(16))) _Float16 v16h;
typedef __attribute__((ext_vector_type(8)))  _Float16 v8h;
typedef __attribute__((ext_vector_type(4)))  _Float16 v4h;
typedef __attribute__((ext_vector_type(8)))  float    v8f;

#define IMIN(a, b) ((a) < (b) ? (a) : (b))

// k-permutation: swap k in [8,16) with [16,24) so a lane's 16 fragment halves
// {hi..hi+7, 16+hi..16+hi+7} are contiguous in LDS. Blocks of 8 move whole.
__device__ __forceinline__ int kperm(int c) {
  return ((((c >> 3) ^ (c >> 4)) & 1) ? (c ^ 24) : c);
}

__device__ __forceinline__ v16h load_frag(const _Float16* p) {
  union { v16h v; v8h h[2]; } u;
  u.h[0] = *(const v8h*)p;
  u.h[1] = *(const v8h*)(p + 8);
  return u.v;
}

// Build an A-operand fragment from two separated 8-half groups.
__device__ __forceinline__ v16h load_frag2(const _Float16* p0,
                                           const _Float16* p1) {
  union { v16h v; v8h h[2]; } u;
  u.h[0] = *(const v8h*)p0;
  u.h[1] = *(const v8h*)p1;
  return u.v;
}

__device__ __forceinline__ float half16_max(float x) {
  x = fmaxf(x, __shfl_xor(x, 1, 32));
  x = fmaxf(x, __shfl_xor(x, 2, 32));
  x = fmaxf(x, __shfl_xor(x, 4, 32));
  x = fmaxf(x, __shfl_xor(x, 8, 32));
  return x;
}
__device__ __forceinline__ float half16_sum(float x) {
  x += __shfl_xor(x, 1, 32);
  x += __shfl_xor(x, 2, 32);
  x += __shfl_xor(x, 4, 32);
  x += __shfl_xor(x, 8, 32);
  return x;
}

// ---------------------------------------------------------------------------
// LayerNorm: f32 in, f16 out (matmul operand precision). One block per row.
// ---------------------------------------------------------------------------
__global__ __launch_bounds__(256) void ln_kernel(
    const float* __restrict__ x, const float* __restrict__ w,
    const float* __restrict__ b, _Float16* __restrict__ y, int C) {
  __shared__ float sred[16];
  int row = blockIdx.x;
  const float* xr = x + (size_t)row * C;
  _Float16* yr = y + (size_t)row * C;
  float s = 0.f, s2 = 0.f;
  for (int i = threadIdx.x; i < C; i += 256) {
    float v = xr[i];
    s += v;
    s2 += v * v;
  }
  for (int m = 16; m >= 1; m >>= 1) {
    s += __shfl_xor(s, m, 32);
    s2 += __shfl_xor(s2, m, 32);
  }
  int wv = threadIdx.x >> 5;
  if ((threadIdx.x & 31) == 0) {
    sred[wv] = s;
    sred[8 + wv] = s2;
  }
  __syncthreads();
  if (threadIdx.x < 32) {
    float a = (threadIdx.x < 8) ? sred[threadIdx.x] : 0.f;
    float a2 = (threadIdx.x < 8) ? sred[8 + threadIdx.x] : 0.f;
    for (int m = 4; m >= 1; m >>= 1) {
      a += __shfl_xor(a, m, 32);
      a2 += __shfl_xor(a2, m, 32);
    }
    if (threadIdx.x == 0) {
      sred[0] = a;
      sred[8] = a2;
    }
  }
  __syncthreads();
  float mean = sred[0] / (float)C;
  float var = sred[8] / (float)C - mean * mean;
  float rstd = rsqrtf(var + 1e-5f);
  for (int i = threadIdx.x; i < C; i += 256) {
    yr[i] = (_Float16)((xr[i] - mean) * rstd * w[i] + b[i]);
  }
}

__global__ __launch_bounds__(256) void cvt_half_kernel(
    const float* __restrict__ x, _Float16* __restrict__ y, int n) {
  int i = blockIdx.x * 256 + threadIdx.x;
  if (i < n) y[i] = (_Float16)x[i];
}

// ---------------------------------------------------------------------------
// WMMA GEMM, double-buffered. A is f16 [M,K]; W is f32 [K,N].
// If Y16 != null: write f16 output (no bias/res). Else: f32 out + bias + res.
// 256 thr = 8 waves, block tile 128x128, K-step 32.
// ---------------------------------------------------------------------------
__global__ __launch_bounds__(256) void gemm_kernel(
    const _Float16* __restrict__ A, const float* __restrict__ W,
    const float* __restrict__ bias, const float* __restrict__ res,
    float* __restrict__ Y, _Float16* __restrict__ Y16, int M, int N, int K) {
  __shared__ __align__(16) _Float16 As[2][128][40];
  __shared__ __align__(16) _Float16 Bt[2][128][40];
  int tid = threadIdx.x;
  int wave = tid >> 5;
  int lane = tid & 31;
  int l16 = lane & 15;
  int hi8 = (lane >> 4) << 3;
  int hi2 = (lane >> 4) << 4;
  int bm = blockIdx.y * 128;
  int bn = blockIdx.x * 128;

  v8f acc[8] = {};
  int nk = K >> 5;

  // Prologue: stage tile 0 into buffer 0.
#pragma unroll
  for (int i = 0; i < 2; ++i) {
    int e8 = tid + 256 * i;       // 0..511 : (row, 8-half chunk)
    int r = e8 >> 2;
    int c8 = (e8 & 3) << 3;
    int gr = bm + r;
    v8h av = {};
    if (gr < M) av = *(const v8h*)(A + (size_t)gr * K + c8);
    *(v8h*)&As[0][r][kperm(c8)] = av;
  }
#pragma unroll
  for (int i = 0; i < 4; ++i) {
    int e = tid + 256 * i;
    int c = e & 127;
    int r4 = (e >> 7) << 2;
    const float* wp = W + (size_t)r4 * N + bn + c;
    v4h h4;
    h4[0] = (_Float16)wp[0];
    h4[1] = (_Float16)wp[(size_t)N];
    h4[2] = (_Float16)wp[2 * (size_t)N];
    h4[3] = (_Float16)wp[3 * (size_t)N];
    *(v4h*)&Bt[0][c][kperm(r4)] = h4;
  }
  __syncthreads();

  for (int kt = 0; kt < nk; ++kt) {
    int cur = kt & 1;
    bool have_next = (kt + 1) < nk;
    int k0n = (kt + 1) << 5;

    // Issue next tile's global loads into registers (overlaps WMMA chain).
    v8h areg[2];
    float breg[4][4];
    if (have_next) {
#pragma unroll
      for (int i = 0; i < 2; ++i) {
        int e8 = tid + 256 * i;
        int r = e8 >> 2;
        int c8 = (e8 & 3) << 3;
        int gr = bm + r;
        areg[i] = (v8h){};
        if (gr < M) {
          const _Float16* ap = A + (size_t)gr * K + k0n + c8;
          areg[i] = *(const v8h*)ap;
          if (k0n + 32 < K) __builtin_prefetch(ap + 32, 0, 1);
        }
      }
#pragma unroll
      for (int i = 0; i < 4; ++i) {
        int e = tid + 256 * i;
        int c = e & 127;
        int r4 = (e >> 7) << 2;
        const float* wp = W + (size_t)(k0n + r4) * N + bn + c;
        breg[i][0] = wp[0];
        breg[i][1] = wp[(size_t)N];
        breg[i][2] = wp[2 * (size_t)N];
        breg[i][3] = wp[3 * (size_t)N];
        if (k0n + 32 < K) __builtin_prefetch(wp + 32 * (size_t)N, 0, 1);
      }
    }

    // Compute from current buffer.
    v16h af = load_frag(&As[cur][wave * 16 + l16][hi2]);
    {
      v16h b0 = load_frag(&Bt[cur][l16][hi2]);
      v16h b1 = load_frag(&Bt[cur][16 + l16][hi2]);
      v16h b2 = load_frag(&Bt[cur][32 + l16][hi2]);
      v16h b3 = load_frag(&Bt[cur][48 + l16][hi2]);
      acc[0] = __builtin_amdgcn_wmma_f32_16x16x32_f16(false, af, false, b0,
                                                      (short)0, acc[0], false, false);
      acc[1] = __builtin_amdgcn_wmma_f32_16x16x32_f16(false, af, false, b1,
                                                      (short)0, acc[1], false, false);
      acc[2] = __builtin_amdgcn_wmma_f32_16x16x32_f16(false, af, false, b2,
                                                      (short)0, acc[2], false, false);
      acc[3] = __builtin_amdgcn_wmma_f32_16x16x32_f16(false, af, false, b3,
                                                      (short)0, acc[3], false, false);
    }
    {
      v16h b4 = load_frag(&Bt[cur][64 + l16][hi2]);
      v16h b5 = load_frag(&Bt[cur][80 + l16][hi2]);
      v16h b6 = load_frag(&Bt[cur][96 + l16][hi2]);
      v16h b7 = load_frag(&Bt[cur][112 + l16][hi2]);
      acc[4] = __builtin_amdgcn_wmma_f32_16x16x32_f16(false, af, false, b4,
                                                      (short)0, acc[4], false, false);
      acc[5] = __builtin_amdgcn_wmma_f32_16x16x32_f16(false, af, false, b5,
                                                      (short)0, acc[5], false, false);
      acc[6] = __builtin_amdgcn_wmma_f32_16x16x32_f16(false, af, false, b6,
                                                      (short)0, acc[6], false, false);
      acc[7] = __builtin_amdgcn_wmma_f32_16x16x32_f16(false, af, false, b7,
                                                      (short)0, acc[7], false, false);
    }

    // Store next tile into the other buffer.
    if (have_next) {
      int nxt = cur ^ 1;
#pragma unroll
      for (int i = 0; i < 2; ++i) {
        int e8 = tid + 256 * i;
        int r = e8 >> 2;
        int c8 = (e8 & 3) << 3;
        *(v8h*)&As[nxt][r][kperm(c8)] = areg[i];
      }
#pragma unroll
      for (int i = 0; i < 4; ++i) {
        int e = tid + 256 * i;
        int c = e & 127;
        int r4 = (e >> 7) << 2;
        v4h h4;
        h4[0] = (_Float16)breg[i][0];
        h4[1] = (_Float16)breg[i][1];
        h4[2] = (_Float16)breg[i][2];
        h4[3] = (_Float16)breg[i][3];
        *(v4h*)&Bt[nxt][c][kperm(r4)] = h4;
      }
    }
    __syncthreads();
  }

  if (Y16) {
#pragma unroll
    for (int t = 0; t < 8; ++t) {
      int n = bn + t * 16 + l16;
#pragma unroll
      for (int r = 0; r < 8; ++r) {
        int m = bm + wave * 16 + r + hi8;
        if (m < M) Y16[(size_t)m * N + n] = (_Float16)acc[t][r];
      }
    }
  } else {
#pragma unroll
    for (int t = 0; t < 8; ++t) {
      int n = bn + t * 16 + l16;
      float bv = bias ? bias[n] : 0.f;
#pragma unroll
      for (int r = 0; r < 8; ++r) {
        int m = bm + wave * 16 + r + hi8;
        if (m < M) {
          size_t idx = (size_t)m * N + n;
          float val = acc[t][r] + bv;
          if (res) val += res[idx];
          Y[idx] = val;
        }
      }
    }
  }
}

// ---------------------------------------------------------------------------
// Fused GEMM + GEGLU: inner16 = (A@W[:, :5120]+bp) * gelu(A@W[:, 5120:]+bg).
// A f16, W f32, inner f16. Never materializes the 335MB pg tensor.
// ---------------------------------------------------------------------------
__global__ __launch_bounds__(256) void gemm_geglu_kernel(
    const _Float16* __restrict__ A, const float* __restrict__ W,
    const float* __restrict__ bias, _Float16* __restrict__ inner,
    int M, int K) {
  const int NI = 5120;
  const int NW = 10240;
  __shared__ __align__(16) _Float16 As[128][40];
  __shared__ __align__(16) _Float16 Bp[128][40];
  __shared__ __align__(16) _Float16 Bg[128][40];
  int tid = threadIdx.x;
  int wave = tid >> 5;
  int lane = tid & 31;
  int l16 = lane & 15;
  int hi8 = (lane >> 4) << 3;
  int hi2 = (lane >> 4) << 4;
  int bm = blockIdx.y * 128;
  int bn = blockIdx.x * 128;

  v8f accp[8] = {};
  v8f accg[8] = {};

  for (int k0 = 0; k0 < K; k0 += 32) {
#pragma unroll
    for (int i = 0; i < 2; ++i) {
      int e8 = tid + 256 * i;
      int r = e8 >> 2;
      int c8 = (e8 & 3) << 3;
      int gr = bm + r;
      v8h av = {};
      if (gr < M) av = *(const v8h*)(A + (size_t)gr * K + k0 + c8);
      *(v8h*)&As[r][kperm(c8)] = av;
    }
#pragma unroll
    for (int i = 0; i < 4; ++i) {
      int e = tid + 256 * i;
      int c = e & 127;
      int r4 = (e >> 7) << 2;
      const float* wp = W + (size_t)(k0 + r4) * NW + bn + c;
      const float* wg = wp + NI;
      v4h hp, hg;
      hp[0] = (_Float16)wp[0];
      hp[1] = (_Float16)wp[(size_t)NW];
      hp[2] = (_Float16)wp[2 * (size_t)NW];
      hp[3] = (_Float16)wp[3 * (size_t)NW];
      hg[0] = (_Float16)wg[0];
      hg[1] = (_Float16)wg[(size_t)NW];
      hg[2] = (_Float16)wg[2 * (size_t)NW];
      hg[3] = (_Float16)wg[3 * (size_t)NW];
      if (k0 + 32 < K) __builtin_prefetch(wp + 32 * (size_t)NW, 0, 1);
      *(v4h*)&Bp[c][kperm(r4)] = hp;
      *(v4h*)&Bg[c][kperm(r4)] = hg;
    }
    __syncthreads();

    v16h af = load_frag(&As[wave * 16 + l16][hi2]);
#pragma unroll
    for (int t = 0; t < 8; t += 2) {
      v16h bp0 = load_frag(&Bp[t * 16 + l16][hi2]);
      v16h bg0 = load_frag(&Bg[t * 16 + l16][hi2]);
      v16h bp1 = load_frag(&Bp[(t + 1) * 16 + l16][hi2]);
      v16h bg1 = load_frag(&Bg[(t + 1) * 16 + l16][hi2]);
      accp[t] = __builtin_amdgcn_wmma_f32_16x16x32_f16(
          false, af, false, bp0, (short)0, accp[t], false, false);
      accg[t] = __builtin_amdgcn_wmma_f32_16x16x32_f16(
          false, af, false, bg0, (short)0, accg[t], false, false);
      accp[t + 1] = __builtin_amdgcn_wmma_f32_16x16x32_f16(
          false, af, false, bp1, (short)0, accp[t + 1], false, false);
      accg[t + 1] = __builtin_amdgcn_wmma_f32_16x16x32_f16(
          false, af, false, bg1, (short)0, accg[t + 1], false, false);
    }
    __syncthreads();
  }

#pragma unroll
  for (int t = 0; t < 8; ++t) {
    int n = bn + t * 16 + l16;
    float bp = bias[n];
    float bg = bias[NI + n];
#pragma unroll
    for (int r = 0; r < 8; ++r) {
      int m = bm + wave * 16 + r + hi8;
      if (m < M) {
        float p = accp[t][r] + bp;
        float g = accg[t][r] + bg;
        float ge = 0.5f * g * (1.f + erff(g * 0.70710678118654752f));
        inner[(size_t)m * NI + n] = (_Float16)(p * ge);
      }
    }
  }
}

// ---------------------------------------------------------------------------
// Flash-style WMMA attention, all-f16 operands. mode 0: dense K/V with
// per-seq stride kss. mode 1: sparse-causal (virtual Sk=512: rows <256 from
// frame 0, >=256 from frame max(f-1,0)). grid.x = nSeq*8, grid.y = Sq/(16*nw).
// ---------------------------------------------------------------------------
__global__ __launch_bounds__(128) void attn_kernel(
    const _Float16* __restrict__ Q, const _Float16* __restrict__ Kb,
    const _Float16* __restrict__ Vb, _Float16* __restrict__ O,
    int Sq, int Sk, long long qss, long long kss, float scale, int mode) {
  __shared__ __align__(16) _Float16 Ks[32][168];
  __shared__ __align__(16) _Float16 Vt[160][40];
  __shared__ __align__(16) float sP[4][16][36];
  int tid = threadIdx.x;
  int nthreads = blockDim.x;
  int nwaves = nthreads >> 5;
  int wave = tid >> 5;
  int lane = tid & 31;
  int l16 = lane & 15;
  int hi8 = (lane >> 4) << 3;
  int hi2 = (lane >> 4) << 4;
  int bh = blockIdx.x;
  int seq = bh >> 3, head = bh & 7;
  const _Float16* qb = Q + (size_t)seq * qss + head * 160;
  _Float16* ob = O + (size_t)seq * qss + head * 160;
  const _Float16* kb0 = Kb + head * 160;
  const _Float16* vb0 = Vb + head * 160;
  size_t seqoff = (size_t)seq * kss;
  int bseq16 = (seq >> 4) << 4;
  int fprev = ((seq & 15) > 0) ? (seq & 15) - 1 : 0;
  auto kvrow = [&](int kv) -> size_t {
    kv = IMIN(kv, Sk - 1);
    if (mode == 0) return seqoff + (size_t)kv * 1280;
    int part = kv >> 8, tok = kv & 255;
    int srcf = part ? fprev : 0;
    return ((size_t)(bseq16 + srcf) * 256 + tok) * 1280;
  };
  int qm = (blockIdx.y * nwaves + wave) << 4;

  // Per-wave Q fragments straight from global f16 (no conversion).
  v16h qf[5];
  {
    const _Float16* qp = qb + (size_t)(qm + l16) * 1280;
#pragma unroll
    for (int c = 0; c < 5; ++c)
      qf[c] = load_frag2(qp + c * 32 + hi8, qp + c * 32 + 16 + hi8);
  }

  v8f acc[10] = {};
  float mrow[8], lrow[8];
#pragma unroll
  for (int r = 0; r < 8; ++r) {
    mrow[r] = -1e30f;
    lrow[r] = 0.f;
  }

  int nkv = (Sk + 31) >> 5;
  for (int it = 0; it < nkv; ++it) {
    int kv0 = it << 5;
    __syncthreads();
    // Stage K tile (32 kv x 160 ch): pure v8h copies, permuted blocks.
    for (int e8 = tid; e8 < 32 * 20; e8 += nthreads) {
      int r = e8 / 20;
      int c8 = (e8 % 20) << 3;
      v8h kv8 = *(const v8h*)(kb0 + kvrow(kv0 + r) + c8);
      *(v8h*)&Ks[r][(c8 & ~31) + kperm(c8 & 31)] = kv8;
    }
    // Stage V tile transposed -> Vt[ch][kperm(kv%32)].
    for (int e = tid; e < 160 * 8; e += nthreads) {
      int ch = e % 160;
      int r4 = (e / 160) << 2;
      v4h h4;
      h4[0] = vb0[kvrow(kv0 + r4 + 0) + ch];
      h4[1] = vb0[kvrow(kv0 + r4 + 1) + ch];
      h4[2] = vb0[kvrow(kv0 + r4 + 2) + ch];
      h4[3] = vb0[kvrow(kv0 + r4 + 3) + ch];
      *(v4h*)&Vt[ch][kperm(r4)] = h4;
    }
    __syncthreads();

    // ---- S = Q @ K^T ----
    v8f sa = {}, sb = {};
#pragma unroll
    for (int c = 0; c < 5; ++c) {
      v16h kfa = load_frag(&Ks[l16][c * 32 + hi2]);
      v16h kfb = load_frag(&Ks[16 + l16][c * 32 + hi2]);
      sa = __builtin_amdgcn_wmma_f32_16x16x32_f16(false, qf[c], false, kfa,
                                                  (short)0, sa, false, false);
      sb = __builtin_amdgcn_wmma_f32_16x16x32_f16(false, qf[c], false, kfb,
                                                  (short)0, sb, false, false);
    }
    // ---- online softmax ----
    bool maA = (kv0 + l16) < Sk;
    bool maB = (kv0 + 16 + l16) < Sk;
    float newm[8];
#pragma unroll
    for (int r = 0; r < 8; ++r) {
      float va = maA ? sa[r] * scale : -1e30f;
      float vbx = maB ? sb[r] * scale : -1e30f;
      sa[r] = va;
      sb[r] = vbx;
      float mx = half16_max(fmaxf(va, vbx));
      newm[r] = fmaxf(mrow[r], mx);
    }
#pragma unroll
    for (int r = 0; r < 8; ++r) {
      float alpha = __expf(mrow[r] - newm[r]);
      mrow[r] = newm[r];
      lrow[r] *= alpha;
#pragma unroll
      for (int t = 0; t < 10; ++t) acc[t][r] *= alpha;
      float pa = __expf(sa[r] - newm[r]);
      float pb = __expf(sb[r] - newm[r]);
      sa[r] = pa;
      sb[r] = pb;
      lrow[r] += half16_sum(pa + pb);
    }
    // ---- P -> fragment-ordered staging ----
#pragma unroll
    for (int r = 0; r < 8; ++r) {
      sP[wave][r + hi8][kperm(l16)] = sa[r];
      sP[wave][r + hi8][kperm(16 + l16)] = sb[r];
    }
    __syncthreads();
    v16h pf;
    {
      const float* pp = &sP[wave][l16][hi2];
#pragma unroll
      for (int j = 0; j < 16; ++j) pf[j] = (_Float16)pp[j];
    }
    // ---- O += P @ V, fragment loads paired ----
#pragma unroll
    for (int t = 0; t < 10; t += 2) {
      v16h vf0 = load_frag(&Vt[t * 16 + l16][hi2]);
      v16h vf1 = load_frag(&Vt[(t + 1) * 16 + l16][hi2]);
      acc[t] = __builtin_amdgcn_wmma_f32_16x16x32_f16(false, pf, false, vf0,
                                                      (short)0, acc[t], false,
                                                      false);
      acc[t + 1] = __builtin_amdgcn_wmma_f32_16x16x32_f16(
          false, pf, false, vf1, (short)0, acc[t + 1], false, false);
    }
  }

#pragma unroll
  for (int t = 0; t < 10; ++t) {
    int n = t * 16 + l16;
#pragma unroll
    for (int r = 0; r < 8; ++r) {
      int m = qm + r + hi8;
      ob[(size_t)m * 1280 + n] = (_Float16)(acc[t][r] / lrow[r]);
    }
  }
}

// ---------------------------------------------------------------------------
// Transposes between [b*f, 256, 1280] and [b*256, 16, 1280] (f32, float4).
// ---------------------------------------------------------------------------
__global__ __launch_bounds__(256) void to_temporal_kernel(
    const float* __restrict__ h, float* __restrict__ ht, int total4) {
  int i = blockIdx.x * 256 + threadIdx.x;
  if (i >= total4) return;
  int col4 = i % 320;
  int row = i / 320;  // ht row: (b*256 + d)*16 + f
  int f = row & 15;
  int bd = row >> 4;
  int d = bd & 255;
  int b = bd >> 8;
  int srow = (b * 16 + f) * 256 + d;
  ((float4*)ht)[i] = ((const float4*)h)[(size_t)srow * 320 + col4];
}

__global__ __launch_bounds__(256) void from_temporal_kernel(
    const float* __restrict__ ht, float* __restrict__ out, int total4) {
  int i = blockIdx.x * 256 + threadIdx.x;
  if (i >= total4) return;
  int col4 = i % 320;
  int row = i / 320;  // out row: (b*16 + f)*256 + d
  int d = row & 255;
  int bfi = row >> 8;
  int f = bfi & 15;
  int b = bfi >> 4;
  int srow = (b * 256 + d) * 16 + f;
  ((float4*)out)[i] = ((const float4*)ht)[(size_t)srow * 320 + col4];
}

// ---------------------------------------------------------------------------
// Host orchestration.
// ---------------------------------------------------------------------------
extern "C" void kernel_launch(void* const* d_in, const int* in_sizes, int n_in,
                              void* d_out, int out_size, void* d_ws,
                              size_t ws_size, hipStream_t stream) {
  const float* hid = (const float*)d_in[0];
  const float* enc = (const float*)d_in[1];
  const float* n1w = (const float*)d_in[2];
  const float* n1b = (const float*)d_in[3];
  const float* a1wq = (const float*)d_in[4];
  const float* a1wk = (const float*)d_in[5];
  const float* a1wv = (const float*)d_in[6];
  const float* a1wo = (const float*)d_in[7];
  const float* a1bo = (const float*)d_in[8];
  const float* n2w = (const float*)d_in[9];
  const float* n2b = (const float*)d_in[10];
  const float* a2wq = (const float*)d_in[11];
  const float* a2wk = (const float*)d_in[12];
  const float* a2wv = (const float*)d_in[13];
  const float* a2wo = (const float*)d_in[14];
  const float* a2bo = (const float*)d_in[15];
  const float* n3w = (const float*)d_in[16];
  const float* n3b = (const float*)d_in[17];
  const float* ffw1 = (const float*)d_in[18];
  const float* ffb1 = (const float*)d_in[19];
  const float* ffw2 = (const float*)d_in[20];
  const float* ffb2 = (const float*)d_in[21];
  const float* ntw = (const float*)d_in[22];
  const float* ntb = (const float*)d_in[23];
  const float* atwq = (const float*)d_in[24];
  const float* atwk = (const float*)d_in[25];
  const float* atwv = (const float*)d_in[26];
  const float* atwo = (const float*)d_in[27];
  const float* atbo = (const float*)d_in[28];

  const int R = 8192;
  const size_t RC = (size_t)R * 1280;
  char* wsb = (char*)d_ws;
  float* h  = (float*)wsb;                        // f32 residual stream
  float* ht = (float*)(wsb + RC * 4);             // f32 temporal residual
  _Float16* nh    = (_Float16*)(wsb + RC * 8);    // f16 intermediates
  _Float16* q16   = nh + RC;
  _Float16* k16   = q16 + RC;
  _Float16* v16   = k16 + RC;
  _Float16* ao    = v16 + RC;
  _Float16* inner = ao + RC;                      // 4*RC halves
  _Float16* enc16 = inner + 4 * RC;               // 2464*768
  _Float16* enck  = enc16 + (size_t)2464 * 768;   // 32*77*1280
  _Float16* encv  = enck + (size_t)32 * 77 * 1280;

  const float scale = 0.07905694150420949f;  // 160^-0.5

  auto gemm16 = [&](const _Float16* A, const float* W, _Float16* Y16, int M,
                    int N, int K) {
    dim3 grid(N / 128, (M + 127) / 128);
    gemm_kernel<<<grid, 256, 0, stream>>>(A, W, nullptr, nullptr, nullptr,
                                          Y16, M, N, K);
  };
  auto gemm32 = [&](const _Float16* A, const float* W, const float* bias,
                    const float* res, float* Y, int M, int N, int K) {
    dim3 grid(N / 128, (M + 127) / 128);
    gemm_kernel<<<grid, 256, 0, stream>>>(A, W, bias, res, Y, nullptr,
                                          M, N, K);
  };

  // Encoder hidden states -> f16 (once).
  {
    int n = 2464 * 768;
    cvt_half_kernel<<<(n + 255) / 256, 256, 0, stream>>>(enc, enc16, n);
  }

  // ---- attn1: sparse-causal self attention (gather fused in staging) ----
  ln_kernel<<<R, 256, 0, stream>>>(hid, n1w, n1b, nh, 1280);
  gemm16(nh, a1wq, q16, R, 1280, 1280);
  gemm16(nh, a1wk, k16, R, 1280, 1280);
  gemm16(nh, a1wv, v16, R, 1280, 1280);
  {
    dim3 ga(32 * 8, 4);  // 4 waves * 16 rows = 64 q rows per block
    attn_kernel<<<ga, 128, 0, stream>>>(q16, k16, v16, ao, 256, 512,
                                        256LL * 1280, 256LL * 1280, scale,
                                        /*mode=*/1);
  }
  gemm32(ao, a1wo, a1bo, hid, h, R, 1280, 1280);

  // ---- attn2: cross attention ----
  ln_kernel<<<R, 256, 0, stream>>>(h, n2w, n2b, nh, 1280);
  gemm16(nh, a2wq, q16, R, 1280, 1280);
  gemm16(enc16, a2wk, enck, 32 * 77, 1280, 768);
  gemm16(enc16, a2wv, encv, 32 * 77, 1280, 768);
  {
    dim3 ga(32 * 8, 4);
    attn_kernel<<<ga, 128, 0, stream>>>(q16, enck, encv, ao, 256, 77,
                                        256LL * 1280, 77LL * 1280, scale,
                                        /*mode=*/0);
  }
  gemm32(ao, a2wo, a2bo, h, h, R, 1280, 1280);

  // ---- GEGLU FFN ----
  ln_kernel<<<R, 256, 0, stream>>>(h, n3w, n3b, nh, 1280);
  {
    dim3 grid(5120 / 128, R / 128);
    gemm_geglu_kernel<<<grid, 256, 0, stream>>>(nh, ffw1, ffb1, inner, R, 1280);
  }
  gemm32(inner, ffw2, ffb2, h, h, R, 1280, 5120);

  // ---- temporal self attention ----
  {
    int t4 = R * 320;
    to_temporal_kernel<<<(t4 + 255) / 256, 256, 0, stream>>>(h, ht, t4);
  }
  ln_kernel<<<R, 256, 0, stream>>>(ht, ntw, ntb, nh, 1280);
  gemm16(nh, atwq, q16, R, 1280, 1280);
  gemm16(nh, atwk, k16, R, 1280, 1280);
  gemm16(nh, atwv, v16, R, 1280, 1280);
  {
    dim3 ga(512 * 8, 1);  // one wave per block: Sq = 16
    attn_kernel<<<ga, 32, 0, stream>>>(q16, k16, v16, ao, 16, 16, 16LL * 1280,
                                       16LL * 1280, scale, /*mode=*/0);
  }
  gemm32(ao, atwo, atbo, ht, h, R, 1280, 1280);  // temporal out -> h (f32)
  {
    int t4 = R * 320;
    from_temporal_kernel<<<(t4 + 255) / 256, 256, 0, stream>>>(
        h, (float*)d_out, t4);
  }
}
